// DETR_78847009619970
// MI455X (gfx1250) — compile-verified, matrix operand order checked by reading, not run on hardware
//
#include <hip/hip_runtime.h>
#include <hip/hip_bf16.h>

typedef __attribute__((ext_vector_type(16))) _Float16 v16h;
typedef __attribute__((ext_vector_type(8)))  _Float16 v8h;
typedef __attribute__((ext_vector_type(8)))  float    v8f;

#define BB_    64
#define PP_    150
#define TT_    150
#define CC_    81
#define PROWS  160     // P padded to 10 tiles of 16 (zero rows 150..159)
#define KPAD   96      // 81 padded to 3 k-steps of 32 (zero cols 81..95)
#define IMGSZ  320.0f
#define EPSF   1e-07f
#define NOOBJ  80

// ---------------------------------------------------------------------------
// Kernel 0: neglogp[b, p, k] = (p<150 && k<81) ? -log(prob[b,p,k]) : 0, in f16.
// Padding rows/cols with zeros removes ALL guards from the WMMA A-operand path.
// ---------------------------------------------------------------------------
__global__ __launch_bounds__(256)
void neglogp_kernel(const float* __restrict__ prob, _Float16* __restrict__ nlp)
{
    const int idx = blockIdx.x * 256 + threadIdx.x;   // over B*PROWS*KPAD
    if (idx >= BB_ * PROWS * KPAD) return;
    const int k = idx % KPAD;
    const int p = (idx / KPAD) % PROWS;
    const int b = idx / (KPAD * PROWS);
    float v = 0.0f;
    if (p < PP_ && k < CC_) v = -__logf(prob[((size_t)b * PP_ + p) * CC_ + k]);
    nlp[idx] = (_Float16)v;
}

// ---------------------------------------------------------------------------
// Kernel A: cost matrix. One wave32 per (batch, 16x16 tile).
// class_nll = (-logP)[16x96] x OneHot[96x16] via v_wmma_f32_16x16x32_f16.
// A fragment per k-step = two unconditional 16-byte global_load_b128 per lane:
//   slots 0..7  <- K = ks*32 + 8*h + (0..7)      (contiguous)
//   slots 8..15 <- K = ks*32 + 16 + 8*h + (0..7) (contiguous)
// Predicted boxes for the row-tile staged once in LDS.
// ---------------------------------------------------------------------------
__global__ __launch_bounds__(32)
void cost_tile_kernel(const _Float16* __restrict__ nlp, // [B,PROWS,KPAD]
                      const float* __restrict__ pbox,   // [B,P,4]
                      const float* __restrict__ labels, // [B,T,5]
                      float* __restrict__ cost)         // [B,P,T]
{
    const int tile = blockIdx.x;          // b*100 + mt*10 + nt
    const int b    = tile / 100;
    const int mt   = (tile / 10) % 10;
    const int nt   = tile % 10;
    const int lane = threadIdx.x;
    const int h    = lane >> 4;           // half-wave
    const int lr   = lane & 15;

    // ---- stage the 16 predicted boxes of this row-tile in LDS ----
    __shared__ float sbox[16][4];
    if (lane < 16) {
        const int m = mt * 16 + lane;
        const float* PB = pbox + (size_t)(b * PP_ + (m < PP_ ? m : PP_ - 1)) * 4;
        sbox[lane][0] = PB[0]; sbox[lane][1] = PB[1];
        sbox[lane][2] = PB[2]; sbox[lane][3] = PB[3];
    }
    __syncthreads();

    // ---- label row for this lane's column (clamped; guarded by tcol later) ----
    const int tcol = nt * 16 + lr;
    const float* L = labels + (size_t)(b * TT_ + (tcol < TT_ ? tcol : TT_ - 1)) * 5;
    const float lb0 = L[0] / IMGSZ, lb1 = L[1] / IMGSZ;
    const float lb2 = L[2] / IMGSZ, lb3 = L[3] / IMGSZ;
    const int   lab_cls = (int)L[4];
    const int   cls = (tcol < TT_) ? lab_cls : -1;   // OOB column -> all-zero one-hot

    // ---- WMMA accumulate over K = 3 x 32 ----
    const _Float16* nrow = nlp + ((size_t)b * PROWS + mt * 16 + lr) * KPAD;
    v8f c = {};
    #pragma unroll
    for (int ks = 0; ks < 3; ++ks) {
        const v8h lo = *(const v8h*)(nrow + ks * 32 + 8 * h);
        const v8h hi = *(const v8h*)(nrow + ks * 32 + 16 + 8 * h);
        v16h a, bmat;
        #pragma unroll
        for (int j = 0; j < 8; ++j) { a[j] = lo[j]; a[j + 8] = hi[j]; }
        #pragma unroll
        for (int j = 0; j < 16; ++j) {
            const int K = ks * 32 + j + 8 * h + (j >= 8 ? 8 : 0);
            bmat[j] = (_Float16)((K == cls) ? 1.0f : 0.0f);
        }
        c = __builtin_amdgcn_wmma_f32_16x16x32_f16(
                /*neg_a=*/false, a, /*neg_b=*/false, bmat,
                /*c_mod=*/(short)0, c, /*reuse_a=*/false, /*reuse_b=*/false);
    }

    // ---- bbox terms for the 8 C/D elements this lane holds ----
    // C/D layout: register r -> M = mt*16 + r + 8*h, N = lane%16
    const int n_global = tcol;
    const float lx0 = lb0 - 0.5f * lb2, ly0 = lb1 - 0.5f * lb3;
    const float lx1 = lb0 + 0.5f * lb2, ly1 = lb1 + 0.5f * lb3;
    const float area_l = (lx1 - lx0) * (ly1 - ly0);
    const float mask   = (lab_cls != NOOBJ) ? 1.0f : 0.0f;

    #pragma unroll
    for (int r = 0; r < 8; ++r) {
        const int mrow = r + 8 * h;             // 0..15 within tile
        const int m_global = mt * 16 + mrow;
        if (m_global >= PP_ || n_global >= TT_) continue;
        const float pcx = sbox[mrow][0], pcy = sbox[mrow][1];
        const float pw  = sbox[mrow][2], ph  = sbox[mrow][3];
        const float l1 = 0.25f * (fabsf(pcx - lb0) + fabsf(pcy - lb1) +
                                  fabsf(pw  - lb2) + fabsf(ph  - lb3));
        const float px0 = pcx - 0.5f * pw, py0 = pcy - 0.5f * ph;
        const float px1 = pcx + 0.5f * pw, py1 = pcy + 0.5f * ph;
        const float ix0 = fmaxf(px0, lx0), iy0 = fmaxf(py0, ly0);
        const float ix1 = fminf(px1, lx1), iy1 = fminf(py1, ly1);
        const float inter = fmaxf(ix1 - ix0, 0.f) * fmaxf(iy1 - iy0, 0.f);
        const float area_p = (px1 - px0) * (py1 - py0);
        const float uni  = area_p + area_l - inter;
        const float iou  = inter / (uni + EPSF);
        const float cx0 = fminf(px0, lx0), cy0 = fminf(py0, ly0);
        const float cx1 = fmaxf(px1, lx1), cy1 = fmaxf(py1, ly1);
        const float enc = fmaxf(cx1 - cx0, 0.f) * fmaxf(cy1 - cy0, 0.f);
        const float giou = iou - (enc - uni) / (enc + EPSF);
        const float val  = c[r] + (5.0f * l1 + 2.0f * (1.0f - giou)) * mask;
        cost[((size_t)b * PP_ + m_global) * TT_ + n_global] = val;
    }
}

// ---------------------------------------------------------------------------
// Kernel B: S[p,t] = sum_b cost[b,p,t]
// ---------------------------------------------------------------------------
__global__ __launch_bounds__(256)
void batch_sum_kernel(const float* __restrict__ cost, float* __restrict__ S)
{
    const int idx = blockIdx.x * blockDim.x + threadIdx.x;
    if (idx >= PP_ * TT_) return;
    float s = 0.0f;
    for (int b = 0; b < BB_; ++b) s += cost[(size_t)b * PP_ * TT_ + idx];
    S[idx] = s;
}

// ---------------------------------------------------------------------------
// Kernel C: Jonker-Volgenant LSA, one wave32 per batch, float64 state in LDS
// (matches the reference's float64 Hungarian; argmin ties -> smallest column).
// ---------------------------------------------------------------------------
__global__ __launch_bounds__(32)
void hungarian_kernel(const float* __restrict__ cost, int* __restrict__ match)
{
    const int b = blockIdx.x;
    const float* Cb = cost + (size_t)b * PP_ * TT_;
    const int lane = threadIdx.x;

    __shared__ double u[PP_ + 1];
    __shared__ double v[TT_ + 1];
    __shared__ double minv[TT_ + 1];
    __shared__ int    way[TT_ + 1];
    __shared__ int    pcol[TT_ + 1];
    __shared__ int    used[TT_ + 1];
    __shared__ double redv[32];
    __shared__ int    redj[32];
    __shared__ int    sh_j0;

    for (int j = lane; j <= TT_; j += 32) { v[j] = 0.0; pcol[j] = 0; way[j] = 0; }
    for (int i = lane; i <= PP_; i += 32) u[i] = 0.0;
    __syncthreads();

    const double INFD = 1e300;

    for (int i = 1; i <= PP_; ++i) {
        if (lane == 0) { pcol[0] = i; sh_j0 = 0; }
        for (int j = lane; j <= TT_; j += 32) { minv[j] = INFD; used[j] = 0; }
        __syncthreads();

        int j1 = 0;
        while (true) {
            const int j0 = sh_j0;
            if (lane == 0) used[j0] = 1;
            __syncthreads();
            const int    i0   = pcol[j0];
            const double u_i0 = u[i0];
            const float* row  = Cb + (size_t)(i0 - 1) * TT_;
            __builtin_prefetch(row, 0, 1);   // global_prefetch_b8

            double bestv = INFD;
            int    bestj = TT_ + 1;
            for (int j = lane + 1; j <= TT_; j += 32) {
                if (!used[j]) {
                    const double cur = (double)row[j - 1] - u_i0 - v[j];
                    if (cur < minv[j]) { minv[j] = cur; way[j] = j0; }
                    const double mv = minv[j];
                    if (mv < bestv) { bestv = mv; bestj = j; }   // ascending j -> first min
                }
            }
            redv[lane] = bestv; redj[lane] = bestj;
            __syncthreads();
            if (lane == 0) {
                for (int k = 1; k < 32; ++k) {
                    if (redv[k] < redv[0] ||
                        (redv[k] == redv[0] && redj[k] < redj[0])) {
                        redv[0] = redv[k]; redj[0] = redj[k];
                    }
                }
                sh_j0 = redj[0];
            }
            __syncthreads();
            const double delta = redv[0];
            j1 = sh_j0;
            for (int j = lane; j <= TT_; j += 32) {
                if (used[j]) { u[pcol[j]] += delta; v[j] -= delta; }  // distinct rows
                else         { minv[j] -= delta; }
            }
            __syncthreads();
            if (pcol[j1] == 0) break;
        }
        if (lane == 0) {                    // augment along alternating path
            int j0 = j1;
            while (j0 != 0) { const int jw = way[j0]; pcol[j0] = pcol[jw]; j0 = jw; }
        }
        __syncthreads();
    }

    for (int j = lane + 1; j <= TT_; j += 32)
        match[b * TT_ + (j - 1)] = pcol[j] - 1;     // row assigned to column j-1
}

// ---------------------------------------------------------------------------
// Kernel D: loss = (1/(B*T)) * sum_i sum_t S[match_i[t], t]
// ---------------------------------------------------------------------------
__global__ __launch_bounds__(256)
void loss_kernel(const float* __restrict__ S, const int* __restrict__ match,
                 float* __restrict__ out)
{
    __shared__ float red[256];
    float s = 0.0f;
    for (int idx = threadIdx.x; idx < BB_ * TT_; idx += 256) {
        const int bb = idx / TT_, t = idx % TT_;
        const int r = match[bb * TT_ + t];
        s += S[r * TT_ + t];
    }
    red[threadIdx.x] = s;
    __syncthreads();
    for (int off = 128; off > 0; off >>= 1) {
        if ((int)threadIdx.x < off) red[threadIdx.x] += red[threadIdx.x + off];
        __syncthreads();
    }
    if (threadIdx.x == 0) out[0] = red[0] / (float)(BB_ * TT_);
}

// ---------------------------------------------------------------------------
extern "C" void kernel_launch(void* const* d_in, const int* in_sizes, int n_in,
                              void* d_out, int out_size, void* d_ws, size_t ws_size,
                              hipStream_t stream)
{
    (void)in_sizes; (void)n_in; (void)out_size; (void)ws_size;
    const float* prob   = (const float*)d_in[0];   // [64,150,81]
    const float* pbox   = (const float*)d_in[1];   // [64,150,4]
    const float* labels = (const float*)d_in[2];   // [64,150,5]

    const size_t NLP_ELEMS  = (size_t)BB_ * PROWS * KPAD;       // 983,040 f16
    const size_t NLP_BYTES  = NLP_ELEMS * sizeof(_Float16);     // 1,966,080 B (16B aligned)

    _Float16* nlp  = (_Float16*)d_ws;
    float*    cost = (float*)((char*)d_ws + NLP_BYTES);         // 1,440,000 f32
    float*    S    = cost + (size_t)BB_ * PP_ * TT_;            //    22,500 f32
    int*      match = (int*)(S + PP_ * TT_);                    //     9,600 i32
    float*    out  = (float*)d_out;

    neglogp_kernel<<<dim3((BB_ * PROWS * KPAD + 255) / 256), dim3(256), 0, stream>>>(prob, nlp);
    cost_tile_kernel<<<dim3(BB_ * 100), dim3(32), 0, stream>>>(nlp, pbox, labels, cost);
    batch_sum_kernel<<<dim3((PP_ * TT_ + 255) / 256), dim3(256), 0, stream>>>(cost, S);
    hungarian_kernel<<<dim3(BB_), dim3(32), 0, stream>>>(cost, match);
    loss_kernel<<<dim3(1), dim3(256), 0, stream>>>(S, match, out);
}
